// MultiHeadAttention_62740882260325
// MI455X (gfx1250) — compile-verified, matrix operand order checked by reading, not run on hardware
//
#include <hip/hip_runtime.h>
#include <hip/hip_bf16.h>
#include <stdint.h>

#define S_LEN 2048
#define EMB   2048
#define NH    16
#define HD    128

typedef __attribute__((ext_vector_type(16))) __bf16 v16bf;
typedef __attribute__((ext_vector_type(8)))  __bf16 v8bf;
typedef __attribute__((ext_vector_type(8)))  float  v8f;
typedef __attribute__((ext_vector_type(4)))  unsigned int u32x4;
typedef __attribute__((ext_vector_type(8)))  int          i32x8;
typedef __attribute__((ext_vector_type(4)))  int          i32x4;

union FragU { v16bf f; struct { v8bf lo, hi; } p; };

// Load one 16x32 bf16 WMMA operand fragment from global (A-layout; B uses the
// same per-lane pattern with "row" = B column, both K-contiguous in memory).
// Per ISA 7.12.2: lanes 0-15 hold K=0..7 (v0-3) and K=16..23 (v4-7),
// lanes 16-31 hold K=8..15 and K=24..31.
static __device__ inline v16bf load_frag(const __bf16* __restrict__ base, int ld, int lane) {
    const int l  = lane & 15;
    const int hi = lane >> 4;
    const __bf16* p = base + (size_t)l * ld + hi * 8;
    FragU u;
    u.p.lo = *(const v8bf*)(p);
    u.p.hi = *(const v8bf*)(p + 16);
    return u.f;
}

static __device__ inline v8f wmma_bf16(v16bf a, v16bf b, v8f c) {
    return __builtin_amdgcn_wmma_f32_16x16x32_bf16(false, a, false, b, (short)0, c, false, false);
}

// ---------------------------------------------------------------------- TDM
// Issue one Tensor-Data-Mover 2D tile load of (tileRows x tileCols) bf16
// elements from a row-major [rows, ld] bf16 tensor into LDS at byte offset
// ldsOff, appending LDS padding after each row (pad codes per D# §8.4:
// padInterval: 0=2,1=4,..,7=256 DWORDs between pads; padAmount: n+1 DWORDs)
// so ds_load_b128 fragment fetches are bank-conflict-free.
static __device__ inline void tdm_load_tile(const __bf16* gtile, unsigned ldsOff,
                                            int rows, int ld,
                                            int tileCols, int tileRows,
                                            unsigned padInterval, unsigned padAmount) {
    const unsigned long long ga = (unsigned long long)(uintptr_t)gtile;
    u32x4 g0;
    g0.x = 1u;                                      // count=1, user descriptor
    g0.y = ldsOff;                                  // lds_addr (bytes)
    g0.z = (unsigned)ga;                            // global_addr[31:0]
    g0.w = (unsigned)((ga >> 32) & 0x01FFFFFFu)     // global_addr[56:32]
         | (2u << 30);                              // type = 2 ("image")
    i32x8 g1;
    g1[0] = (int)((1u << 16)                        // data_size: 2 bytes
                | (1u << 20)                        // pad_enable
                | (padInterval << 22)
                | (padAmount << 25));
    g1[1] = (int)(((unsigned)ld & 0xFFFFu) << 16);             // tensor_dim0 lo
    g1[2] = (int)((((unsigned)ld >> 16) & 0xFFFFu)
                | (((unsigned)rows & 0xFFFFu) << 16));          // dim0 hi | dim1 lo
    g1[3] = (int)(((((unsigned)rows) >> 16) & 0xFFFFu)
                | (((unsigned)tileCols) << 16));                // dim1 hi | tile_dim0
    g1[4] = tileRows;                                           // tile_dim1
    g1[5] = ld;                                                 // tensor_dim0_stride
    g1[6] = 0;
    g1[7] = 0;
    const i32x4 z4 = (i32x4){0, 0, 0, 0};
#if defined(__clang_major__) && (__clang_major__ >= 23)
    const i32x8 z8 = (i32x8){0, 0, 0, 0, 0, 0, 0, 0};
    __builtin_amdgcn_tensor_load_to_lds(g0, g1, z4, z4, z8, 0);
#else
    __builtin_amdgcn_tensor_load_to_lds(g0, g1, z4, z4, 0);
#endif
}

// ---------------------------------------------------------------- fp32 -> bf16
__global__ __launch_bounds__(256) void cvt_bf16_kernel(const float* __restrict__ src,
                                                       __bf16* __restrict__ dst, int n) {
    for (int i = blockIdx.x * blockDim.x + threadIdx.x; i < n; i += gridDim.x * blockDim.x)
        dst[i] = (__bf16)src[i];
}

// ------------------------------------------------- C = A[M,K] * W[N,K]^T + bias
// TDM-staged, double-buffered LDS pipeline; 8 waves compute a 128x128 C tile.
// outMode: 0 = f32 row-major, 1 = bf16 row-major, 2 = bf16 transposed [N,M]
#define LDS_PITCH 40   // bf16 elems per 32-elem LDS tile row (+8 TDM pad)
#define KT_PITCH  136  // bf16 elems per 128-elem LDS tile row (+8 TDM pad)

#define GEMM_LDS_FRAG(tile, row0, dst) do {                              \
        FragU _u;                                                        \
        const int _r = (row0) + l;                                       \
        _u.p.lo = *(const v8bf*)&(tile)[_r * LDS_PITCH + hi * 8];        \
        _u.p.hi = *(const v8bf*)&(tile)[_r * LDS_PITCH + hi * 8 + 16];   \
        (dst) = _u.f; } while (0)

__global__ __launch_bounds__(256) void gemm_bf16_kernel(
    const __bf16* __restrict__ A, const __bf16* __restrict__ W,
    const float* __restrict__ bias,
    float* __restrict__ outF, __bf16* __restrict__ outB,
    int M, int N, int K, int outMode)
{
    __shared__ alignas(16) __bf16 Atile[2][128 * LDS_PITCH];
    __shared__ alignas(16) __bf16 Btile[2][128 * LDS_PITCH];

    const int lane = threadIdx.x & 31;
    const int w    = threadIdx.x >> 5;            // 8 waves: 4 (M) x 2 (N)
    const int bm   = blockIdx.y * 128;
    const int bn   = blockIdx.x * 128;
    const int wmr  = (w >> 1) * 32;               // wave row offset in block tile
    const int wnc  = (w & 1) * 64;                // wave col offset in block tile
    const int l    = lane & 15;
    const int hi   = lane >> 4;

    v8f acc[2][4];
#pragma unroll
    for (int i = 0; i < 2; ++i)
#pragma unroll
        for (int j = 0; j < 4; ++j)
            acc[i][j] = (v8f){0.f,0.f,0.f,0.f,0.f,0.f,0.f,0.f};

    // Prologue: stage k-step 0 (wave 0 issues; EXEC-independent DMA).
    if (threadIdx.x < 32) {
        tdm_load_tile(A + (size_t)bm * K, (unsigned)(uintptr_t)&Atile[0][0], M, K, 32, 128, 3, 3);
        tdm_load_tile(W + (size_t)bn * K, (unsigned)(uintptr_t)&Btile[0][0], N, K, 32, 128, 3, 3);
    }

    const int nk = K / 32;
    for (int i = 0; i < nk; ++i) {
        const int cur = i & 1;
        if (threadIdx.x < 32) {
            if (i + 1 < nk) {
                const int k0n = (i + 1) * 32;
                tdm_load_tile(A + (size_t)bm * K + k0n,
                              (unsigned)(uintptr_t)&Atile[cur ^ 1][0], M, K, 32, 128, 3, 3);
                tdm_load_tile(W + (size_t)bn * K + k0n,
                              (unsigned)(uintptr_t)&Btile[cur ^ 1][0], N, K, 32, 128, 3, 3);
                __builtin_amdgcn_s_wait_tensorcnt(2);   // step i done; step i+1 in flight
            } else {
                __builtin_amdgcn_s_wait_tensorcnt(0);
            }
        }
        __syncthreads();

        v16bf a0, a1, bfr[4];
        GEMM_LDS_FRAG(Atile[cur], wmr,      a0);
        GEMM_LDS_FRAG(Atile[cur], wmr + 16, a1);
#pragma unroll
        for (int j = 0; j < 4; ++j)
            GEMM_LDS_FRAG(Btile[cur], wnc + j * 16, bfr[j]);
#pragma unroll
        for (int j = 0; j < 4; ++j) {
            acc[0][j] = wmma_bf16(a0, bfr[j], acc[0][j]);
            acc[1][j] = wmma_bf16(a1, bfr[j], acc[1][j]);
        }
        __syncthreads();
    }

    const int wm = bm + wmr;
    const int wn = bn + wnc;
    if (outMode == 0) {
#pragma unroll
        for (int i = 0; i < 2; ++i)
#pragma unroll
            for (int j = 0; j < 4; ++j) {
                const int col = wn + j * 16 + l;
                const float bc = bias ? bias[col] : 0.f;
#pragma unroll
                for (int r = 0; r < 8; ++r)
                    outF[(size_t)(wm + i * 16 + r + hi * 8) * N + col] = acc[i][j][r] + bc;
            }
    } else if (outMode == 1) {
#pragma unroll
        for (int i = 0; i < 2; ++i)
#pragma unroll
            for (int j = 0; j < 4; ++j) {
                const int col = wn + j * 16 + l;
                const float bc = bias ? bias[col] : 0.f;
#pragma unroll
                for (int r = 0; r < 8; ++r)
                    outB[(size_t)(wm + i * 16 + r + hi * 8) * N + col] =
                        (__bf16)(acc[i][j][r] + bc);
            }
    } else {
#pragma unroll
        for (int i = 0; i < 2; ++i)
#pragma unroll
            for (int j = 0; j < 4; ++j) {
                const int col = wn + j * 16 + l;
                const float bc = bias ? bias[col] : 0.f;
#pragma unroll
                for (int r = 0; r < 8; ++r)
                    outB[(size_t)col * M + (wm + i * 16 + r + hi * 8)] =
                        (__bf16)(acc[i][j][r] + bc);
            }
    }
}

// ----------------------------------------------------------- flash attention
// Q, K: bf16 [S, EMB] (head h in cols h*HD..); Vt: bf16 [EMB, S] (row h*HD+d).
// Block = 8 waves x 16 query rows, one head. Per key-step of 32, wave 0 TDM-
// stages the shared K tile (32xHD) and Vt tile (HDx32) into double-buffered
// LDS; all waves consume via ds_load_b128.
__global__ __launch_bounds__(256) void flash_attn_kernel(
    const __bf16* __restrict__ Q, const __bf16* __restrict__ Kmat,
    const __bf16* __restrict__ Vt, __bf16* __restrict__ Ao)
{
    __shared__ alignas(16) __bf16 Ktile[2][32 * KT_PITCH];     // rows t, cols d
    __shared__ alignas(16) __bf16 Vtile[2][128 * LDS_PITCH];   // rows d, cols t
    __shared__ alignas(16) __bf16 Plds[8][16 * 32];            // wave-private P

    const int lane = threadIdx.x & 31;
    const int w    = threadIdx.x >> 5;
    const int q0   = blockIdx.x * 128 + w * 16;
    const int h    = blockIdx.y;
    const int l    = lane & 15;
    const int hi   = lane >> 4;
    const float sc = 0.08838834764831845f;    // 1/sqrt(HD)

    v16bf qf[4];
#pragma unroll
    for (int kk = 0; kk < 4; ++kk)
        qf[kk] = load_frag(Q + (size_t)q0 * EMB + h * HD + kk * 32, EMB, lane);

    v8f acc[8];
#pragma unroll
    for (int d = 0; d < 8; ++d) acc[d] = (v8f){0.f,0.f,0.f,0.f,0.f,0.f,0.f,0.f};
    float m[8], lsum[8];
#pragma unroll
    for (int r = 0; r < 8; ++r) { m[r] = -3.0e38f; lsum[r] = 0.f; }

    // Prologue: stage key-step 0.
    if (threadIdx.x < 32) {
        tdm_load_tile(Kmat + (size_t)0 * EMB + h * HD,
                      (unsigned)(uintptr_t)&Ktile[0][0], S_LEN, EMB, 128, 32, 5, 3);
        tdm_load_tile(Vt + (size_t)(h * HD) * S_LEN + 0,
                      (unsigned)(uintptr_t)&Vtile[0][0], EMB, S_LEN, 32, 128, 3, 3);
    }

    const int nt = S_LEN / 32;
    for (int i = 0; i < nt; ++i) {
        const int cur = i & 1;
        if (threadIdx.x < 32) {
            if (i + 1 < nt) {
                const int t1 = (i + 1) * 32;
                tdm_load_tile(Kmat + (size_t)t1 * EMB + h * HD,
                              (unsigned)(uintptr_t)&Ktile[cur ^ 1][0], S_LEN, EMB, 128, 32, 5, 3);
                tdm_load_tile(Vt + (size_t)(h * HD) * S_LEN + t1,
                              (unsigned)(uintptr_t)&Vtile[cur ^ 1][0], EMB, S_LEN, 32, 128, 3, 3);
                __builtin_amdgcn_s_wait_tensorcnt(2);
            } else {
                __builtin_amdgcn_s_wait_tensorcnt(0);
            }
        }
        __syncthreads();

        // ---- scores: Q (16xHD) x K^T (HDx32) -> two 16x16 f32 tiles
        v8f s0 = (v8f){0.f,0.f,0.f,0.f,0.f,0.f,0.f,0.f};
        v8f s1 = (v8f){0.f,0.f,0.f,0.f,0.f,0.f,0.f,0.f};
#pragma unroll
        for (int kk = 0; kk < 4; ++kk) {
            v16bf b0, b1;
            {
                FragU _u;
                const int r0 = l;          // B column t = K-tile row (lanes 0..15)
                _u.p.lo = *(const v8bf*)&Ktile[cur][r0 * KT_PITCH + kk * 32 + hi * 8];
                _u.p.hi = *(const v8bf*)&Ktile[cur][r0 * KT_PITCH + kk * 32 + hi * 8 + 16];
                b0 = _u.f;
                const int r1 = 16 + l;
                _u.p.lo = *(const v8bf*)&Ktile[cur][r1 * KT_PITCH + kk * 32 + hi * 8];
                _u.p.hi = *(const v8bf*)&Ktile[cur][r1 * KT_PITCH + kk * 32 + hi * 8 + 16];
                b1 = _u.f;
            }
            s0 = wmma_bf16(qf[kk], b0, s0);
            s1 = wmma_bf16(qf[kk], b1, s1);
        }
        // ---- online softmax (rows live across 16 lanes of a half-wave)
        float corr[8];
#pragma unroll
        for (int r = 0; r < 8; ++r) {
            s0[r] *= sc; s1[r] *= sc;
            float mx = fmaxf(s0[r], s1[r]);
            mx = fmaxf(mx, __shfl_xor(mx, 1));
            mx = fmaxf(mx, __shfl_xor(mx, 2));
            mx = fmaxf(mx, __shfl_xor(mx, 4));
            mx = fmaxf(mx, __shfl_xor(mx, 8));
            const float mn = fmaxf(m[r], mx);
            corr[r] = __expf(m[r] - mn);
            m[r] = mn;
            s0[r] = __expf(s0[r] - mn);
            s1[r] = __expf(s1[r] - mn);
            float sum = s0[r] + s1[r];
            sum += __shfl_xor(sum, 1);
            sum += __shfl_xor(sum, 2);
            sum += __shfl_xor(sum, 4);
            sum += __shfl_xor(sum, 8);
            lsum[r] = lsum[r] * corr[r] + sum;
        }
#pragma unroll
        for (int d = 0; d < 8; ++d)
#pragma unroll
            for (int r = 0; r < 8; ++r) acc[d][r] *= corr[r];

        // ---- P: C-layout -> A-layout transpose via wave-private LDS tile.
        // DS ops from the same wave are in-order; no cross-wave sharing.
#pragma unroll
        for (int r = 0; r < 8; ++r) {
            const int row = r + hi * 8;
            Plds[w][row * 32 + l]      = (__bf16)s0[r];
            Plds[w][row * 32 + 16 + l] = (__bf16)s1[r];
        }
        FragU pu;
        pu.p.lo = *(const v8bf*)&Plds[w][l * 32 + hi * 8];
        pu.p.hi = *(const v8bf*)&Plds[w][l * 32 + hi * 8 + 16];

        // ---- acc += P (16x32) x V (32xHD); Vt-tile rows are contiguous over t
#pragma unroll
        for (int d = 0; d < 8; ++d) {
            v16bf bv;
            GEMM_LDS_FRAG(Vtile[cur], d * 16, bv);
            acc[d] = wmma_bf16(pu.f, bv, acc[d]);
        }
        __syncthreads();
    }

    // ---- normalize and emit bf16 attention output [S, EMB]
#pragma unroll
    for (int r = 0; r < 8; ++r) lsum[r] = 1.f / lsum[r];
#pragma unroll
    for (int d = 0; d < 8; ++d) {
        const int col = h * HD + d * 16 + l;
#pragma unroll
        for (int r = 0; r < 8; ++r) {
            const int row = q0 + r + hi * 8;
            Ao[(size_t)row * EMB + col] = (__bf16)(acc[d][r] * lsum[r]);
        }
    }
}

extern "C" void kernel_launch(void* const* d_in, const int* in_sizes, int n_in,
                              void* d_out, int out_size, void* d_ws, size_t ws_size,
                              hipStream_t stream) {
    (void)in_sizes; (void)n_in; (void)out_size; (void)ws_size;
    const float* x  = (const float*)d_in[0];
    const float* Wq = (const float*)d_in[1];
    const float* bq = (const float*)d_in[2];
    const float* Wk = (const float*)d_in[3];
    const float* bk = (const float*)d_in[4];
    const float* Wv = (const float*)d_in[5];
    const float* bv = (const float*)d_in[6];
    const float* Wo = (const float*)d_in[7];
    const float* bo = (const float*)d_in[8];
    float* out = (float*)d_out;

    char* ws = (char*)d_ws;
    const size_t TB = (size_t)S_LEN * EMB * sizeof(__bf16);   // 8 MB per bf16 tensor
    __bf16* xb  = (__bf16*)(ws + 0 * TB);
    __bf16* Wqb = (__bf16*)(ws + 1 * TB);
    __bf16* Wkb = (__bf16*)(ws + 2 * TB);
    __bf16* Wvb = (__bf16*)(ws + 3 * TB);
    __bf16* Wob = (__bf16*)(ws + 4 * TB);
    __bf16* Qb  = (__bf16*)(ws + 5 * TB);
    __bf16* Kb  = (__bf16*)(ws + 6 * TB);
    __bf16* Vtb = (__bf16*)(ws + 7 * TB);   // transposed [EMB, S]
    __bf16* Ab  = (__bf16*)(ws + 8 * TB);   // attention output bf16 [S, EMB]

    const int nElem = S_LEN * EMB;
    hipLaunchKernelGGL(cvt_bf16_kernel, dim3(2048), dim3(256), 0, stream, x,  xb,  nElem);
    hipLaunchKernelGGL(cvt_bf16_kernel, dim3(2048), dim3(256), 0, stream, Wq, Wqb, nElem);
    hipLaunchKernelGGL(cvt_bf16_kernel, dim3(2048), dim3(256), 0, stream, Wk, Wkb, nElem);
    hipLaunchKernelGGL(cvt_bf16_kernel, dim3(2048), dim3(256), 0, stream, Wv, Wvb, nElem);
    hipLaunchKernelGGL(cvt_bf16_kernel, dim3(2048), dim3(256), 0, stream, Wo, Wob, nElem);

    dim3 gg(EMB / 128, S_LEN / 128);
    // Q, K projections (bf16 row-major out)
    hipLaunchKernelGGL(gemm_bf16_kernel, gg, dim3(256), 0, stream,
                       xb, Wqb, bq, (float*)nullptr, Qb, S_LEN, EMB, EMB, 1);
    hipLaunchKernelGGL(gemm_bf16_kernel, gg, dim3(256), 0, stream,
                       xb, Wkb, bk, (float*)nullptr, Kb, S_LEN, EMB, EMB, 1);
    // V projection stored transposed: Vt[e, t]
    hipLaunchKernelGGL(gemm_bf16_kernel, gg, dim3(256), 0, stream,
                       xb, Wvb, bv, (float*)nullptr, Vtb, S_LEN, EMB, EMB, 2);
    // attention
    hipLaunchKernelGGL(flash_attn_kernel, dim3(S_LEN / 128, NH), dim3(256), 0, stream,
                       Qb, Kb, Vtb, Ab);
    // output projection -> f32 d_out
    hipLaunchKernelGGL(gemm_bf16_kernel, gg, dim3(256), 0, stream,
                       Ab, Wob, bo, out, (__bf16*)nullptr, S_LEN, EMB, EMB, 0);
}
